// BatchedGAT_cat1_40862318854439
// MI455X (gfx1250) — matrix operand order repaired and checked by reading, not compiled, not run on hardware
//
#include <hip/hip_runtime.h>
#include <hip/hip_bf16.h>

typedef _Float16 v4h  __attribute__((ext_vector_type(4)));
typedef _Float16 v8h  __attribute__((ext_vector_type(8)));
typedef _Float16 v16h __attribute__((ext_vector_type(16)));
typedef float    v8f  __attribute__((ext_vector_type(8)));

#define GAT_B    8
#define GAT_N    4096
#define GAT_M    16
#define GAT_F    128
#define GAT_ROWS 32768            // B*N
#define GAT_C    256              // concat channels
#define GAT_GBLK 2048             // ROWS/16
#define GAT_TOT  (GAT_ROWS * GAT_C)
#define GAT_ALPHA 0.2f
#define GAT_EPS   1e-5f
#define ASTRIDE  136              // LDS A row stride in halves (mult of 8 -> 16B aligned)

__device__ __forceinline__ float wave_sum(float v) {
  #pragma unroll
  for (int o = 16; o > 0; o >>= 1) v += __shfl_xor(v, o, 32);
  return v;
}

__device__ __forceinline__ float dot4(float4 a, float4 b) {
  return a.x*b.x + a.y*b.y + a.z*b.z + a.w*b.w;
}

// ---------------------------------------------------------------- kernel 0
// Convert the two 128x128 weight matrices to f16.
__global__ __launch_bounds__(256) void gat_prep(
    const float* __restrict__ wx_f, const float* __restrict__ wn_f,
    _Float16* __restrict__ wx_h, _Float16* __restrict__ wn_h) {
  int i = blockIdx.x * 256 + threadIdx.x;          // 0..32767
  if (i < GAT_F * GAT_F) wx_h[i] = (_Float16)wx_f[i];
  else {
    int j = i - GAT_F * GAT_F;
    wn_h[j] = (_Float16)wn_f[j];
  }
}

// ---------------------------------------------------------------- kernel 1a
// Factorized attention logits: d_self[b,j] = x[b,j].wa_self,
// d_nei[b,j] = x[b,j].wa_nei  (e[b,n,m] = d_self[b,n] + d_nei[b,idx[n,m]]).
// One wave32 per row; also emits x as f16 (WMMA A operand).
__global__ __launch_bounds__(256) void gat_dots(
    const float* __restrict__ x, const float* __restrict__ wa,
    _Float16* __restrict__ xh, float* __restrict__ dself,
    float* __restrict__ dnei) {
  const int tid  = threadIdx.x;
  const int lane = tid & 31;
  const int row  = (blockIdx.x * 256 + tid) >> 5;

  const float4 xi  = ((const float4*)(x + (size_t)row * GAT_F))[lane];
  const float4 was = ((const float4*)(wa))[lane];
  const float4 wan = ((const float4*)(wa + GAT_F))[lane];

  const float ds = wave_sum(dot4(xi, was));
  const float dn = wave_sum(dot4(xi, wan));

  v4h xo = { (_Float16)xi.x, (_Float16)xi.y, (_Float16)xi.z, (_Float16)xi.w };
  ((v4h*)(xh + (size_t)row * GAT_F))[lane] = xo;
  if (lane == 0) { dself[row] = ds; dnei[row] = dn; }
}

// ---------------------------------------------------------------- kernel 1b
// One wave32 per (b,n) row: gather 16 precomputed logits (broadcast loads),
// LeakyReLU + softmax (redundant per lane, zero shuffles), then the single
// rank-16 feature gather h' = att * x_neib. Emits h' as f16.
__global__ __launch_bounds__(256) void gat_attn(
    const float* __restrict__ x, const int* __restrict__ idx,
    const float* __restrict__ dself, const float* __restrict__ dnei,
    _Float16* __restrict__ hph) {
  const int tid  = threadIdx.x;
  const int lane = tid & 31;
  const int row  = (blockIdx.x * 256 + tid) >> 5;       // global (b,n) row
  const int b    = row >> 12;                           // / 4096
  const int n    = row & (GAT_N - 1);
  const float* xb  = x    + (size_t)b * GAT_N * GAT_F;
  const float* dnb = dnei + (size_t)b * GAT_N;

  const float es = dself[row];

  float em[GAT_M];
  int   jn[GAT_M];
  #pragma unroll
  for (int m = 0; m < GAT_M; ++m) {
    const int j = idx[n * GAT_M + m];                   // same addr all lanes
    jn[m] = j;
    float e = es + dnb[j];                              // broadcast gather
    em[m] = (e > 0.0f) ? e : GAT_ALPHA * e;             // LeakyReLU
  }

  float mx = em[0];
  #pragma unroll
  for (int m = 1; m < GAT_M; ++m) mx = fmaxf(mx, em[m]);
  float den = 0.0f;
  #pragma unroll
  for (int m = 0; m < GAT_M; ++m) { em[m] = __expf(em[m] - mx); den += em[m]; }
  const float rden = 1.0f / den;

  float4 acc = make_float4(0.f, 0.f, 0.f, 0.f);
  #pragma unroll
  for (int m = 0; m < GAT_M; ++m) {
    const float4 xn = ((const float4*)(xb + (size_t)jn[m] * GAT_F))[lane];
    const float a = em[m] * rden;
    acc.x += a * xn.x; acc.y += a * xn.y; acc.z += a * xn.z; acc.w += a * xn.w;
  }

  v4h hp = { (_Float16)acc.x, (_Float16)acc.y, (_Float16)acc.z, (_Float16)acc.w };
  ((v4h*)(hph + (size_t)row * GAT_F))[lane] = hp;
}

// ---------------------------------------------------------------- kernel 2
// 16-row tile per block (8 waves). Round 0: h_x = x@Wx^T, round 1: h_nei = h'@Wn^T.
// Wave w owns 16 output channels; 4x v_wmma_f32_16x16x32_f16 per round (K=128).
// Fused: bias, concat in LDS, row L2-normalize (256 ch), ReLU, store, BN partials.
__global__ __launch_bounds__(256) void gat_gemm(
    const _Float16* __restrict__ xh, const _Float16* __restrict__ hph,
    const _Float16* __restrict__ wx, const _Float16* __restrict__ wn,
    const float* __restrict__ bx, const float* __restrict__ bnb,
    float* __restrict__ hnorm, float* __restrict__ bsum, float* __restrict__ bsum2) {
  __shared__ __align__(16) _Float16 ldsA[16 * ASTRIDE];
  __shared__ float ldsH[16][GAT_C];
  __shared__ float red[16][16];
  __shared__ float rinv[16];

  const int tid   = threadIdx.x;
  const int wv    = tid >> 5;
  const int lane  = tid & 31;
  const int row0  = blockIdx.x * 16;

  #pragma unroll
  for (int r = 0; r < 2; ++r) {
    const _Float16* Asrc = (r == 0) ? xh : hph;
    const _Float16* Wsrc = (r == 0) ? wx : wn;
    const float*    bias = (r == 0) ? bx : bnb;

    // stage 16x128 f16 A tile: 256 threads x 8 halves (16B each)
    {
      const int ar = tid >> 4, ac = (tid & 15) << 3;
      *(v8h*)&ldsA[ar * ASTRIDE + ac] =
          *(const v8h*)&Asrc[(size_t)(row0 + ar) * GAT_F + ac];
    }
    __syncthreads();

    const int nloc  = lane & 15;
    const int nglob = wv * 16 + nloc;
    const int m     = lane & 15;
    const int koffA = (lane >= 16) ? 8 : 0;    // A: 16-bit 16x32 per-lane layout
    const int koffB = (lane >= 16) ? 8 : 0;    // B: same 16-bit per-lane K layout as A
    const _Float16* wrow = Wsrc + (size_t)nglob * GAT_F;

    const float bval = bias[nglob];
    v8f c = { bval, bval, bval, bval, bval, bval, bval, bval };

    #pragma unroll
    for (int kb = 0; kb < GAT_F; kb += 32) {
      v8h alo = *(const v8h*)&ldsA[m * ASTRIDE + kb + koffA];
      v8h ahi = *(const v8h*)&ldsA[m * ASTRIDE + kb + koffA + 16];
      v16h a  = __builtin_shufflevector(alo, ahi, 0,1,2,3,4,5,6,7,8,9,10,11,12,13,14,15);
      v8h blo = *(const v8h*)&wrow[kb + koffB];
    v8h bhi = *(const v8h*)&wrow[kb + koffB + 16];
      v16h bm = __builtin_shufflevector(blo, bhi, 0,1,2,3,4,5,6,7,8,9,10,11,12,13,14,15);
      c = __builtin_amdgcn_wmma_f32_16x16x32_f16(false, a, false, bm,
                                                 (short)0, c, false, false);
    }

    // scatter D tile: lane holds fixed n; VGPR v -> M = v (+8 for upper lanes)
    const int mb = (lane >= 16) ? 8 : 0;
    const int cb = r * GAT_F + wv * 16 + nloc;
    #pragma unroll
    for (int v = 0; v < 8; ++v) ldsH[v + mb][cb] = c[v];
    __syncthreads();
  }

  // row L2 norm over 256 channels: 16 threads x 16-ch segments per row
  const int rr  = tid >> 4;
  const int seg = tid & 15;
  {
    float s = 0.0f;
    #pragma unroll
    for (int i = 0; i < 16; ++i) { float v = ldsH[rr][seg * 16 + i]; s += v * v; }
    red[rr][seg] = s;
  }
  __syncthreads();
  if (tid < 16) {
    float t = 0.0f;
    #pragma unroll
    for (int i = 0; i < 16; ++i) t += red[tid][i];
    rinv[tid] = 1.0f / fmaxf(sqrtf(t), 1e-12f);
  }
  __syncthreads();

  {
    const float iv = rinv[rr];
    const size_t grow = (size_t)(row0 + rr);
    #pragma unroll
    for (int i = 0; i < 16; ++i) {
      const int ch = seg * 16 + i;
      float v = fmaxf(ldsH[rr][ch] * iv, 0.0f);        // normalize + ReLU
      ldsH[rr][ch] = v;
      hnorm[grow * GAT_C + ch] = v;
    }
  }
  __syncthreads();

  // deterministic per-block BN partials: thread = channel
  {
    float s1 = 0.0f, s2 = 0.0f;
    #pragma unroll
    for (int r2 = 0; r2 < 16; ++r2) { float v = ldsH[r2][tid]; s1 += v; s2 += v * v; }
    bsum [(size_t)blockIdx.x * GAT_C + tid] = s1;
    bsum2[(size_t)blockIdx.x * GAT_C + tid] = s2;
  }
}

// ---------------------------------------------------------------- kernel 3
// Reduce 2048 block partials per channel (deterministic tree).
__global__ __launch_bounds__(256) void gat_bnred(
    const float* __restrict__ bsum, const float* __restrict__ bsum2,
    float* __restrict__ gs, float* __restrict__ gs2) {
  const int c = blockIdx.x;            // 256 channels
  const int t = threadIdx.x;           // 256 threads
  float s = 0.0f, s2 = 0.0f;
  for (int i = t; i < GAT_GBLK; i += 256) {
    s  += bsum [(size_t)i * GAT_C + c];
    s2 += bsum2[(size_t)i * GAT_C + c];
  }
  __shared__ float r1[256], r2[256];
  r1[t] = s; r2[t] = s2;
  __syncthreads();
  #pragma unroll
  for (int o = 128; o > 0; o >>= 1) {
    if (t < o) { r1[t] += r1[t + o]; r2[t] += r2[t + o]; }
    __syncthreads();
  }
  if (t == 0) { gs[c] = r1[0]; gs2[c] = r2[0]; }
}

// ---------------------------------------------------------------- kernel 4
__global__ __launch_bounds__(256) void gat_bnapply(
    const float* __restrict__ hnorm, const float* __restrict__ gs,
    const float* __restrict__ gs2, const float* __restrict__ gamma,
    const float* __restrict__ beta, float* __restrict__ out) {
  const size_t i = (size_t)blockIdx.x * 256 + threadIdx.x;
  if (i >= (size_t)GAT_TOT) return;
  const int c = (int)(i & (GAT_C - 1));
  const float mean = gs[c]  * (1.0f / GAT_ROWS);
  const float var  = gs2[c] * (1.0f / GAT_ROWS) - mean * mean;
  const float inv  = rsqrtf(var + GAT_EPS);
  out[i] = gamma[c] * (hnorm[i] - mean) * inv + beta[c];
}

// ---------------------------------------------------------------- launch
extern "C" void kernel_launch(void* const* d_in, const int* in_sizes, int n_in,
                              void* d_out, int out_size, void* d_ws, size_t ws_size,
                              hipStream_t stream) {
  const float* x     = (const float*)d_in[0];
  const int*   idx   = (const int*)  d_in[1];
  const float* Wx    = (const float*)d_in[2];
  const float* Wxb   = (const float*)d_in[3];
  const float* Wn    = (const float*)d_in[4];
  const float* Wnb   = (const float*)d_in[5];
  const float* Wa    = (const float*)d_in[6];
  const float* gamma = (const float*)d_in[7];
  const float* beta  = (const float*)d_in[8];
  float* out = (float*)d_out;

  char* ws = (char*)d_ws;
  const size_t o_xh  = 0;                                      // 8 MB  f16 x
  const size_t o_hph = o_xh  + (size_t)GAT_ROWS * GAT_F * 2;   // 8 MB  f16 h'
  const size_t o_hn  = o_hph + (size_t)GAT_ROWS * GAT_F * 2;   // 32 MB f32 normalized
  const size_t o_wx  = o_hn  + (size_t)GAT_ROWS * GAT_C * 4;   // 32 KB f16 Wx
  const size_t o_wn  = o_wx  + (size_t)GAT_F * GAT_F * 2;      // 32 KB f16 Wn
  const size_t o_bs  = o_wn  + (size_t)GAT_F * GAT_F * 2;      // 2 MB partial sums
  const size_t o_bs2 = o_bs  + (size_t)GAT_GBLK * GAT_C * 4;   // 2 MB partial sumsq
  const size_t o_gs  = o_bs2 + (size_t)GAT_GBLK * GAT_C * 4;   // 1 KB
  const size_t o_gs2 = o_gs  + (size_t)GAT_C * 4;              // 1 KB
  const size_t o_ds  = o_gs2 + (size_t)GAT_C * 4;              // 128 KB d_self
  const size_t o_dn  = o_ds  + (size_t)GAT_ROWS * 4;           // 128 KB d_nei

  _Float16* xh   = (_Float16*)(ws + o_xh);
  _Float16* hph  = (_Float16*)(ws + o_hph);
  float*    hn   = (float*)   (ws + o_hn);
  _Float16* wxh  = (_Float16*)(ws + o_wx);
  _Float16* wnh  = (_Float16*)(ws + o_wn);
  float*    bs   = (float*)   (ws + o_bs);
  float*    bs2  = (float*)   (ws + o_bs2);
  float*    gs   = (float*)   (ws + o_gs);
  float*    gs2  = (float*)   (ws + o_gs2);
  float*    dsf  = (float*)   (ws + o_ds);
  float*    dnf  = (float*)   (ws + o_dn);

  gat_prep<<<128, 256, 0, stream>>>(Wx, Wn, wxh, wnh);
  gat_dots<<<GAT_ROWS / 8, 256, 0, stream>>>(x, Wa, xh, dsf, dnf);
  gat_attn<<<GAT_ROWS / 8, 256, 0, stream>>>(x, idx, dsf, dnf, hph);
  gat_gemm<<<GAT_GBLK, 256, 0, stream>>>(xh, hph, wxh, wnh, Wxb, Wnb, hn, bs, bs2);
  gat_bnred<<<GAT_C, 256, 0, stream>>>(bs, bs2, gs, gs2);
  gat_bnapply<<<GAT_TOT / 256, 256, 0, stream>>>(hn, gs, gs2, gamma, beta, out);
}
